// Decoder_recurrent_3444563771485
// MI455X (gfx1250) — matrix operand order repaired and checked by reading, not compiled, not run on hardware
//
#include <hip/hip_runtime.h>
#include <hip/hip_bf16.h>
#include <stdint.h>

// ---------------------------------------------------------------------------
// Persistent 2-layer LSTM decoder for MI455X (gfx1250).
//  * batch-partitioned: 64 persistent WGs x 16 batch rows, zero inter-WG sync
//  * bf16 WMMA (v_wmma_f32_16x16x32_bf16), f32 accumulate
//  * weights converted f32->bf16 once per launch into d_ws (~24.3 MB, L2-resident)
//  * h state ping-pong in LDS as bf16 (A-matrix operand), c state f32 in LDS
// ---------------------------------------------------------------------------

typedef __attribute__((ext_vector_type(16))) __bf16 bf16x16;
typedef __attribute__((ext_vector_type(8)))  __bf16 bf16x8;
typedef __attribute__((ext_vector_type(8)))  float  f32x8;

#define B_TOT   1024
#define T_STEPS 336
#define H_DIM   1024
#define DFF     7
#define QOUT    9
#define BT      16                       // batch rows per workgroup
#define NWG     (B_TOT / BT)             // 64
#define NTHREADS 512
#define NWAVES  (NTHREADS / 32)          // 16 waves (wave32)
#define JT_PER_WAVE ((H_DIM / 16) / NWAVES)  // 4 column-tiles of 16 per wave
#define HS      1032                     // LDS h row stride (bf16 elems), bank-spread
#define CS      1024                     // LDS c row stride (f32 elems)
#define XS      32                       // padded input width (D_IN=8 -> 32)

// workspace layout (ushort elements)
#define N_HH     ((size_t)4096 * 1024)
#define WHH0_OFF ((size_t)0)
#define WIH1_OFF (N_HH)
#define WHH1_OFF (2 * N_HH)
#define WIH0_OFF (3 * N_HH)                          // [4096][32] zero-padded
#define WOUT_OFF (3 * N_HH + (size_t)4096 * 32)      // [16][1024] zero-padded
#define CV_TOTAL (3 * N_HH + (size_t)4096 * 32 + (size_t)16 * 1024)

__device__ __forceinline__ unsigned short f32_to_bf16_bits(float f) {
  unsigned int u = __float_as_uint(f);
  u += 0x7FFFu + ((u >> 16) & 1u);   // round to nearest even
  return (unsigned short)(u >> 16);
}

__device__ __forceinline__ float sigmoidf_(float x) {
  return 1.0f / (1.0f + __expf(-x));
}
__device__ __forceinline__ float tanhf_(float x) {
  float e = __expf(2.0f * x);
  return 1.0f - 2.0f / (e + 1.0f);
}

// A-matrix (16x32 bf16): lanes0-15 M=m,K=k0..k0+15 (contiguous);
// lanes16-31 same M, K=k0+16..k0+31.
__device__ __forceinline__ bf16x16 loadA(const unsigned short* base, int k0,
                                         int stride, int lane) {
  int m  = lane & 15;
  int kh = (lane >> 4) << 4;           // 0 or 16 elements
  const __bf16* p = (const __bf16*)(base + (size_t)m * stride + k0 + kh);
  bf16x16 r;
  ((bf16x8*)&r)[0] = *(const bf16x8*)p;         // K k0+kh   .. +7
  ((bf16x8*)&r)[1] = *(const bf16x8*)(p + 8);   // K k0+kh+8 .. +15
  return r;
}

// B-matrix (32x16 bf16): lane n<16 holds column n, K=k0..k0+15 (contiguous);
// lane n+16 holds K=k0+16..k0+31.  Row-major W[n][k] is directly the K-major
// B operand -> one 32-byte contiguous load per lane.
__device__ __forceinline__ bf16x16 loadB(const unsigned short* W, int n0,
                                         int k0, int ld, int lane) {
  int n    = lane & 15;
  int koff = (lane >> 4) << 4;         // 0 or 16 elements
  const __bf16* p = (const __bf16*)(W + (size_t)(n0 + n) * ld + k0 + koff);
  return *(const bf16x16*)p;           // 32 B -> 2x global_load_b128
}

__device__ __forceinline__ f32x8 wmma_bf16(bf16x16 a, bf16x16 b, f32x8 c) {
  return __builtin_amdgcn_wmma_f32_16x16x32_bf16(false, a, false, b,
                                                 (short)0, c, false, false);
}

// ---------------------------------------------------------------------------
// one-shot weight conversion / padding  f32 -> bf16
// ---------------------------------------------------------------------------
__global__ void cvt_weights(const float* __restrict__ Wih0,
                            const float* __restrict__ Whh0,
                            const float* __restrict__ Wih1,
                            const float* __restrict__ Whh1,
                            const float* __restrict__ Wout,
                            unsigned short* __restrict__ ws) {
  for (size_t i = (size_t)blockIdx.x * blockDim.x + threadIdx.x; i < CV_TOTAL;
       i += (size_t)gridDim.x * blockDim.x) {
    float v;
    if (i < N_HH) {
      v = Whh0[i];
    } else if (i < 2 * N_HH) {
      v = Wih1[i - N_HH];
    } else if (i < 3 * N_HH) {
      v = Whh1[i - 2 * N_HH];
    } else if (i < WOUT_OFF) {
      size_t l = i - WIH0_OFF;
      int n = (int)(l >> 5), k = (int)(l & 31);
      v = (k < 8) ? Wih0[n * 8 + k] : 0.0f;      // pad D_IN 8 -> 32
    } else {
      size_t l = i - WOUT_OFF;
      int r = (int)(l >> 10), k = (int)(l & 1023);
      v = (r < QOUT) ? Wout[r * 1024 + k] : 0.0f; // pad 9 -> 16 rows
    }
    ws[i] = f32_to_bf16_bits(v);
  }
}

// ---------------------------------------------------------------------------
// persistent decoder kernel: one WG = 16 batch rows for all 336 steps
// ---------------------------------------------------------------------------
__global__ __launch_bounds__(NTHREADS, 1)
void lstm_decoder(const float* __restrict__ inp_y,
                  const float* __restrict__ h_init,
                  const float* __restrict__ c_init,
                  const float* __restrict__ ff,
                  const float* __restrict__ b_ih0, const float* __restrict__ b_hh0,
                  const float* __restrict__ b_ih1, const float* __restrict__ b_hh1,
                  const float* __restrict__ b_out,
                  const unsigned short* __restrict__ ws,
                  float* __restrict__ out) {
  extern __shared__ char smem[];
  unsigned short* h0 = (unsigned short*)smem;          // [2][16][HS] bf16
  unsigned short* h1 = h0 + 2 * 16 * HS;               // [2][16][HS]
  unsigned short* xp = h1 + 2 * 16 * HS;               // [16][XS]
  float* c0 = (float*)(xp + 16 * XS);                  // [16][CS]
  float* c1 = c0 + 16 * CS;                            // [16][CS]

  const unsigned short* Whh0 = ws + WHH0_OFF;
  const unsigned short* Wih1 = ws + WIH1_OFF;
  const unsigned short* Whh1 = ws + WHH1_OFF;
  const unsigned short* Wih0 = ws + WIH0_OFF;
  const unsigned short* Wout = ws + WOUT_OFF;

  const int tid = threadIdx.x;
  const int b0  = blockIdx.x * BT;

  // ---- init state from inputs ----
  for (int i = tid; i < 16 * H_DIM; i += NTHREADS) {
    int m = i >> 10, k = i & 1023;
    h0[m * HS + k] = f32_to_bf16_bits(h_init[(size_t)(0 * B_TOT + b0 + m) * H_DIM + k]);
    h1[m * HS + k] = f32_to_bf16_bits(h_init[(size_t)(1 * B_TOT + b0 + m) * H_DIM + k]);
    c0[m * CS + k] = c_init[(size_t)(0 * B_TOT + b0 + m) * H_DIM + k];
    c1[m * CS + k] = c_init[(size_t)(1 * B_TOT + b0 + m) * H_DIM + k];
  }
  for (int i = tid; i < 16 * XS; i += NTHREADS) {
    int m = i >> 5, k = i & 31;
    float v = 0.0f;
    if (k == 0)      v = inp_y[b0 + m];
    else if (k <= DFF) v = ff[((size_t)(b0 + m) * T_STEPS + 0) * DFF + (k - 1)];
    xp[m * XS + k] = f32_to_bf16_bits(v);
  }
  __syncthreads();

  const int wave  = tid >> 5;
  const int lane  = tid & 31;
  const int nn    = lane & 15;          // owned column within 16-wide tile
  const int mbase = (lane >> 4) * 8;    // C-layout: rows mbase..mbase+7
  const int jbase = wave * (JT_PER_WAVE * 16);

  int cur = 0;
  for (int t = 0; t < T_STEPS; ++t) {
    unsigned short* h0c = h0 + cur * 16 * HS;
    unsigned short* h0n = h0 + (1 - cur) * 16 * HS;
    unsigned short* h1c = h1 + cur * 16 * HS;
    unsigned short* h1n = h1 + (1 - cur) * 16 * HS;

    // ================= layer 0 =================
    for (int jt = 0; jt < JT_PER_WAVE; ++jt) {
      const int j0 = jbase + jt * 16;
      f32x8 aI = {0,0,0,0,0,0,0,0}, aF = aI, aG = aI, aO = aI;

      {  // x @ Wih0^T  (K padded to 32)
        bf16x16 a = loadA(xp, 0, XS, lane);
        aI = wmma_bf16(a, loadB(Wih0, 0 * H_DIM + j0, 0, 32, lane), aI);
        aF = wmma_bf16(a, loadB(Wih0, 1 * H_DIM + j0, 0, 32, lane), aF);
        aG = wmma_bf16(a, loadB(Wih0, 2 * H_DIM + j0, 0, 32, lane), aG);
        aO = wmma_bf16(a, loadB(Wih0, 3 * H_DIM + j0, 0, 32, lane), aO);
      }
#pragma unroll 4
      for (int k0 = 0; k0 < H_DIM; k0 += 32) {  // h0 @ Whh0^T
        bf16x16 a = loadA(h0c, k0, HS, lane);
        aI = wmma_bf16(a, loadB(Whh0, 0 * H_DIM + j0, k0, H_DIM, lane), aI);
        aF = wmma_bf16(a, loadB(Whh0, 1 * H_DIM + j0, k0, H_DIM, lane), aF);
        aG = wmma_bf16(a, loadB(Whh0, 2 * H_DIM + j0, k0, H_DIM, lane), aG);
        aO = wmma_bf16(a, loadB(Whh0, 3 * H_DIM + j0, k0, H_DIM, lane), aO);
      }
      const float bi = b_ih0[0 * H_DIM + j0 + nn] + b_hh0[0 * H_DIM + j0 + nn];
      const float bf = b_ih0[1 * H_DIM + j0 + nn] + b_hh0[1 * H_DIM + j0 + nn];
      const float bg = b_ih0[2 * H_DIM + j0 + nn] + b_hh0[2 * H_DIM + j0 + nn];
      const float bo = b_ih0[3 * H_DIM + j0 + nn] + b_hh0[3 * H_DIM + j0 + nn];
#pragma unroll
      for (int r = 0; r < 8; ++r) {
        int m = mbase + r;
        float iv = sigmoidf_(aI[r] + bi);
        float fv = sigmoidf_(aF[r] + bf);
        float gv = tanhf_(aG[r] + bg);
        float ov = sigmoidf_(aO[r] + bo);
        float cc = c0[m * CS + j0 + nn];
        float cn = fv * cc + iv * gv;
        c0[m * CS + j0 + nn] = cn;
        h0n[m * HS + j0 + nn] = f32_to_bf16_bits(ov * tanhf_(cn));
      }
    }
    __syncthreads();

    // ================= layer 1 =================
    for (int jt = 0; jt < JT_PER_WAVE; ++jt) {
      const int j0 = jbase + jt * 16;
      f32x8 aI = {0,0,0,0,0,0,0,0}, aF = aI, aG = aI, aO = aI;
#pragma unroll 4
      for (int k0 = 0; k0 < H_DIM; k0 += 32) {  // h0_new @ Wih1^T
        bf16x16 a = loadA(h0n, k0, HS, lane);
        aI = wmma_bf16(a, loadB(Wih1, 0 * H_DIM + j0, k0, H_DIM, lane), aI);
        aF = wmma_bf16(a, loadB(Wih1, 1 * H_DIM + j0, k0, H_DIM, lane), aF);
        aG = wmma_bf16(a, loadB(Wih1, 2 * H_DIM + j0, k0, H_DIM, lane), aG);
        aO = wmma_bf16(a, loadB(Wih1, 3 * H_DIM + j0, k0, H_DIM, lane), aO);
      }
#pragma unroll 4
      for (int k0 = 0; k0 < H_DIM; k0 += 32) {  // h1 @ Whh1^T
        bf16x16 a = loadA(h1c, k0, HS, lane);
        aI = wmma_bf16(a, loadB(Whh1, 0 * H_DIM + j0, k0, H_DIM, lane), aI);
        aF = wmma_bf16(a, loadB(Whh1, 1 * H_DIM + j0, k0, H_DIM, lane), aF);
        aG = wmma_bf16(a, loadB(Whh1, 2 * H_DIM + j0, k0, H_DIM, lane), aG);
        aO = wmma_bf16(a, loadB(Whh1, 3 * H_DIM + j0, k0, H_DIM, lane), aO);
      }
      const float bi = b_ih1[0 * H_DIM + j0 + nn] + b_hh1[0 * H_DIM + j0 + nn];
      const float bf = b_ih1[1 * H_DIM + j0 + nn] + b_hh1[1 * H_DIM + j0 + nn];
      const float bg = b_ih1[2 * H_DIM + j0 + nn] + b_hh1[2 * H_DIM + j0 + nn];
      const float bo = b_ih1[3 * H_DIM + j0 + nn] + b_hh1[3 * H_DIM + j0 + nn];
#pragma unroll
      for (int r = 0; r < 8; ++r) {
        int m = mbase + r;
        float iv = sigmoidf_(aI[r] + bi);
        float fv = sigmoidf_(aF[r] + bf);
        float gv = tanhf_(aG[r] + bg);
        float ov = sigmoidf_(aO[r] + bo);
        float cc = c1[m * CS + j0 + nn];
        float cn = fv * cc + iv * gv;
        c1[m * CS + j0 + nn] = cn;
        h1n[m * HS + j0 + nn] = f32_to_bf16_bits(ov * tanhf_(cn));
      }
    }
    __syncthreads();

    // ============ output projection + autoregressive feedback ============
    if (wave == 0) {  // uniform per wave -> EXEC all-1s for WMMA
      f32x8 aY = {0,0,0,0,0,0,0,0};
#pragma unroll 4
      for (int k0 = 0; k0 < H_DIM; k0 += 32) {
        bf16x16 a = loadA(h1n, k0, HS, lane);
        aY = wmma_bf16(a, loadB(Wout, 0, k0, H_DIM, lane), aY);
      }
      if (nn < QOUT) {
        float bb = b_out[nn];
#pragma unroll
        for (int r = 0; r < 8; ++r) {
          int m = mbase + r;
          float y = aY[r] + bb;
          out[((size_t)(b0 + m) * T_STEPS + t) * QOUT + nn] = y;
          if (nn == 0) xp[m * XS + 0] = f32_to_bf16_bits(y);  // y_prev feedback
        }
      }
    }
    if (wave == 1 && lane < 16 && (t + 1) < T_STEPS) {  // next future features
      int m = lane;
#pragma unroll
      for (int k = 0; k < DFF; ++k)
        xp[m * XS + 1 + k] =
            f32_to_bf16_bits(ff[((size_t)(b0 + m) * T_STEPS + (t + 1)) * DFF + k]);
    }
    __syncthreads();
    cur ^= 1;
  }
}

extern "C" void kernel_launch(void* const* d_in, const int* in_sizes, int n_in,
                              void* d_out, int out_size, void* d_ws, size_t ws_size,
                              hipStream_t stream) {
  const float* inp_y = (const float*)d_in[2];
  const float* h_in  = (const float*)d_in[3];
  const float* c_in  = (const float*)d_in[4];
  const float* ff    = (const float*)d_in[5];
  const float* Wih0  = (const float*)d_in[9];
  const float* Whh0  = (const float*)d_in[10];
  const float* bih0  = (const float*)d_in[11];
  const float* bhh0  = (const float*)d_in[12];
  const float* Wih1  = (const float*)d_in[13];
  const float* Whh1  = (const float*)d_in[14];
  const float* bih1  = (const float*)d_in[15];
  const float* bhh1  = (const float*)d_in[16];
  const float* Wout  = (const float*)d_in[17];
  const float* bout  = (const float*)d_in[18];
  unsigned short* ws = (unsigned short*)d_ws;   // needs ~24.3 MB
  float* out = (float*)d_out;

  cvt_weights<<<4096, 256, 0, stream>>>(Wih0, Whh0, Wih1, Whh1, Wout, ws);

  size_t shmem = (size_t)(2 * 16 * HS + 2 * 16 * HS + 16 * XS) * sizeof(unsigned short)
               + (size_t)(2 * 16 * CS) * sizeof(float);   // ~258 KB, fits 320 KB WGP LDS
  lstm_decoder<<<NWG, NTHREADS, shmem, stream>>>(
      inp_y, h_in, c_in, ff, bih0, bhh0, bih1, bhh1, bout, ws, out);
}
